// ModulatedConv1d_2551210574410
// MI455X (gfx1250) — matrix operand order repaired and checked by reading, not compiled, not run on hardware
//
#include <hip/hip_runtime.h>
#include <math.h>

// ---------------- problem constants ----------------
#define B_      16
#define CIN     256
#define COUT    256
#define KW      3
#define SD      512
#define DLEN    8192
#define KDIM    (CIN * KW)          // 768, ordered K = k*256 + c
#define EPS_    1e-8f
#define CONV_SCALE (1.0f / 48.0f)   // 1/sqrt(256*3^2)
#define MOD_SCALE  0.044194173824159216f  // 1/sqrt(512)

// ---------------- WMMA types ----------------
typedef __attribute__((ext_vector_type(16))) __bf16 v16bf;
typedef __attribute__((ext_vector_type(8)))  float  v8f;

union BFrag { v16bf v; unsigned int u[8]; };

__device__ __forceinline__ unsigned short f32_to_bf16(float f) {
    unsigned int u = __float_as_uint(f);
    // round-to-nearest-even
    u += 0x7FFFu + ((u >> 16) & 1u);
    return (unsigned short)(u >> 16);
}

// ---------------- P1: s = style @ (mod_w*scale)^T + mod_b ----------------
__global__ void style_mod_kernel(const float* __restrict__ style,
                                 const float* __restrict__ mod_w,
                                 const float* __restrict__ mod_b,
                                 float* __restrict__ s_out) {
    int g = blockIdx.x * blockDim.x + threadIdx.x;
    if (g >= B_ * CIN) return;
    int b = g >> 8, c = g & 255;
    const float* st = style + (size_t)b * SD;
    const float* mw = mod_w + (size_t)c * SD;
    float acc = 0.0f;
    #pragma unroll 4
    for (int t = 0; t < SD; ++t) acc += st[t] * mw[t];
    s_out[g] = acc * MOD_SCALE + mod_b[c];
}

// ---------------- P2: demod[b][o] = rsqrt(sum (CONV_SCALE*w*s)^2 + eps) ----
__global__ void demod_kernel(const float* __restrict__ weight,
                             const float* __restrict__ s_in,
                             float* __restrict__ demod_out) {
    int g = blockIdx.x * blockDim.x + threadIdx.x;
    if (g >= B_ * COUT) return;
    int b = g >> 8, o = g & 255;
    const float* wp = weight + (size_t)o * KDIM;
    const float* sp = s_in + (size_t)b * CIN;
    float acc = 0.0f;
    for (int c = 0; c < CIN; ++c) {
        float sv = sp[c] * CONV_SCALE;
        #pragma unroll
        for (int k = 0; k < KW; ++k) {
            float w = wp[c * KW + k] * sv;
            acc += w * w;
        }
    }
    demod_out[g] = rsqrtf(acc + EPS_);
}

// ---------------- P3: pack bf16 A fragments in WMMA lane layout ------------
// A tile = 16x32 (MxK). Per ISA: lane<16 : M=lane,  dwords 0..3 -> K 0..7,
//                                            dwords 4..7 -> K 16..23
//                      lane>=16: M=lane-16, dwords 0..3 -> K 8..15,
//                                            dwords 4..7 -> K 24..31
// Fragment f = (b*16 + mtile)*24 + ktile; 256 dwords per fragment.
__global__ void pack_a_kernel(const float* __restrict__ weight,
                              const float* __restrict__ s_in,
                              const float* __restrict__ demod_in,
                              unsigned int* __restrict__ aws) {
    int g = blockIdx.x * blockDim.x + threadIdx.x;
    const int total = B_ * 16 * 24 * 256;   // 1,572,864 dwords
    if (g >= total) return;
    int frag = g >> 8;
    int w    = g & 255;
    int lane = w >> 3;
    int j    = w & 7;

    int b  = frag / (16 * 24);
    int r  = frag % (16 * 24);
    int mt = r / 24;
    int kt = r % 24;

    int m    = mt * 16 + (lane & 15);
    int base = ((lane >> 4) << 3) + ((j >> 2) << 4) + ((j & 3) << 1);

    float dm = demod_in[b * COUT + m] * CONV_SCALE;
    unsigned int packed = 0;
    #pragma unroll
    for (int h = 0; h < 2; ++h) {
        int Kg = kt * 32 + base + h;     // K = k*256 + c
        int c  = Kg & 255;
        int kk = Kg >> 8;
        float val = weight[(size_t)m * KDIM + c * KW + kk] * s_in[b * CIN + c] * dm;
        packed |= (unsigned int)f32_to_bf16(val) << (16 * h);
    }
    aws[g] = packed;
}

// ---------------- main: implicit-GEMM modulated conv via bf16 WMMA ---------
#define BLK_N    128
#define BLK_M    128
#define PITCH_DW 20         // 32 bf16 per row (16 dw) + 4 dw pad -> bank spread

__launch_bounds__(128)
__global__ void modconv_wmma_kernel(const float* __restrict__ input,
                                    const unsigned int* __restrict__ aws,
                                    float* __restrict__ out) {
    __shared__ unsigned int ldsB[BLK_N * PITCH_DW];          // 10,240 B
    unsigned short* ldsB16 = (unsigned short*)ldsB;

    const int d0     = blockIdx.x * BLK_N;
    const int mblock = blockIdx.y;
    const int b      = blockIdx.z;
    const int tid    = threadIdx.x;
    const int lane   = tid & 31;
    const int wave   = tid >> 5;
    const int wm     = wave >> 1;        // 2 waves along M
    const int wn     = wave & 1;         // 2 waves along N

    v8f acc[4][4];
    #pragma unroll
    for (int mi = 0; mi < 4; ++mi)
        #pragma unroll
        for (int ni = 0; ni < 4; ++ni)
            acc[mi][ni] = (v8f){0.f,0.f,0.f,0.f,0.f,0.f,0.f,0.f};

    const float* inb = input + (size_t)b * CIN * DLEN;

    for (int kt = 0; kt < 24; ++kt) {
        const int kk    = kt >> 3;            // conv tap 0..2
        const int cbase = (kt & 7) << 5;      // in-channel base of 32-chunk

        // ---- stage B: input[cbase..+32)[d0-1+kk .. ) -> LDS bf16 [d][c] ----
        #pragma unroll
        for (int it = 0; it < 32; ++it) {
            int idx = it * 128 + tid;
            int c = idx >> 7;
            int j = idx & 127;
            int sd = d0 + j + kk - 1;
            float v = 0.0f;
            if ((unsigned)sd < (unsigned)DLEN)
                v = inb[(size_t)(cbase + c) * DLEN + sd];
            ldsB16[j * (PITCH_DW * 2) + c] = f32_to_bf16(v);
        }
        __syncthreads();

        // ---- load A fragments (prepacked, contiguous 32B per lane) ----
        BFrag afr[4], bfr[4];
        #pragma unroll
        for (int mi = 0; mi < 4; ++mi) {
            int mt = mblock * 8 + wm * 4 + mi;
            const unsigned int* ap =
                aws + ((size_t)((b * 16 + mt) * 24 + kt)) * 256 + lane * 8;
            #pragma unroll
            for (int j = 0; j < 8; ++j) afr[mi].u[j] = ap[j];
        }
        // ---- load B fragments from LDS ----
        // lane<16: (K 0..7,16..23, N=lane)  lane>=16: (K 8..15,24..31, N=lane-16)
        #pragma unroll
        for (int ni = 0; ni < 4; ++ni) {
            int row = wn * 64 + ni * 16 + (lane & 15);
            const unsigned int* bp = ldsB + row * PITCH_DW + ((lane >> 4) << 2);
            #pragma unroll
            for (int j = 0; j < 8; ++j) bfr[ni].u[j] = bp[j + ((j >> 2) << 2)];
        }

        // ---- 16 WMMAs: 64x64 per wave per K-step ----
        #pragma unroll
        for (int mi = 0; mi < 4; ++mi)
            #pragma unroll
            for (int ni = 0; ni < 4; ++ni)
                acc[mi][ni] = __builtin_amdgcn_wmma_f32_16x16x32_bf16(
                    false, afr[mi].v, false, bfr[ni].v,
                    (short)0, acc[mi][ni], false, false);

        __syncthreads();
    }

    // ---- store: VGPR r holds M = r + 8*(lane>=16), N = lane&15 ----
    const int hi = lane >> 4;
    const int nn = lane & 15;
    #pragma unroll
    for (int mi = 0; mi < 4; ++mi) {
        #pragma unroll
        for (int ni = 0; ni < 4; ++ni) {
            int mbase = mblock * BLK_M + wm * 64 + mi * 16 + hi * 8;
            int dcol  = d0 + wn * 64 + ni * 16 + nn;
            float* op = out + ((size_t)b * COUT + mbase) * DLEN + dcol;
            #pragma unroll
            for (int r = 0; r < 8; ++r)
                op[(size_t)r * DLEN] = acc[mi][ni][r];
        }
    }
}

// ---------------- launch ----------------
extern "C" void kernel_launch(void* const* d_in, const int* in_sizes, int n_in,
                              void* d_out, int out_size, void* d_ws, size_t ws_size,
                              hipStream_t stream) {
    const float* input  = (const float*)d_in[0];   // (16,256,8192)
    const float* style  = (const float*)d_in[1];   // (16,512)
    const float* weight = (const float*)d_in[2];   // (256,256,3)
    const float* mod_w  = (const float*)d_in[3];   // (256,512)
    const float* mod_b  = (const float*)d_in[4];   // (256,)
    float* out = (float*)d_out;                    // (16,256,8192)

    char* ws = (char*)d_ws;
    float*        s_buf = (float*)ws;                       // 16 KB
    float*        demod = (float*)(ws + 16384);             // 16 KB
    unsigned int* aws   = (unsigned int*)(ws + 32768);      // 6 MB packed A

    style_mod_kernel<<<(B_ * CIN + 255) / 256, 256, 0, stream>>>(
        style, mod_w, mod_b, s_buf);
    demod_kernel<<<(B_ * COUT + 255) / 256, 256, 0, stream>>>(
        weight, s_buf, demod);
    pack_a_kernel<<<(B_ * 16 * 24 * 256) / 256, 256, 0, stream>>>(
        weight, s_buf, demod, aws);

    dim3 grid(DLEN / BLK_N, COUT / BLK_M, B_);   // 64 x 2 x 16
    modconv_wmma_kernel<<<grid, 128, 0, stream>>>(input, aws, out);
}
